// DynamicNetwork_2637109920279
// MI455X (gfx1250) — matrix operand order repaired and checked by reading, not compiled, not run on hardware
//
#include <hip/hip_runtime.h>

// ---------------------------------------------------------------------------
// DLRM-ish DNAS network on MI455X (gfx1250): wave32 + WMMA f16->f32.
// Pipeline: prep (weight cvt/transpose + alpha scalars + norm zero)
//        -> embed (gather + scale, f16 node01, sumsq norms)
//        -> 4x { relu-GEMM (concat-K), Wc-GEMM (+bias, alpha out scale) }
//        -> classifier GEMV + regs.
// GEMM: 256x64 block tile, 8 waves x (32x64) each, k-step 32, double-buffered
// LDS staged with global_load_async_to_lds_b128 (ASYNCcnt) on the identity-
// alpha fast path; 8 v_wmma_f32_16x16x32_f16 per k-step per wave.
// ---------------------------------------------------------------------------

typedef __attribute__((ext_vector_type(16))) _Float16 v16h;
typedef __attribute__((ext_vector_type(8)))  _Float16 h8;
typedef __attribute__((ext_vector_type(8)))  float    v8f;

#define B_    8192
#define ND_   13
#define NS_   26
#define D_    16
#define VPF_  50000
#define BIN_  400
#define BOUT_ 400

// workspace offsets (bytes), all 256-aligned
#define OFF_NORM  ((size_t)0)                         // 2 floats
#define OFF_SC    ((size_t)256)                       // 4 blocks * 6 floats
#define OFF_W0T   ((size_t)512)                       // 400 x 640 f16
#define OFF_W1T   (OFF_W0T + (size_t)400*640*2)       // 400 x 832 f16
#define OFF_WCT   (OFF_W1T + (size_t)400*832*2)       // 4 x 400 x 416 f16
#define OFF_N01   (OFF_WCT + (size_t)4*400*416*2)     // 8192 x 624 f16
#define OFF_H     (OFF_N01 + (size_t)B_*624*2)        // 8192 x 400 f16
#define OFF_NODES (OFF_H   + (size_t)B_*400*2)        // 8192 x 1600 f16
#define WS_END    (OFF_NODES + (size_t)B_*1600*2)     // ~45.5 MB total

__device__ inline h8 h8zero() {
    h8 v;
#pragma unroll
    for (int j = 0; j < 8; ++j) v[j] = (_Float16)0.0f;
    return v;
}

// async copy of 16B global -> LDS (per lane), tracked by ASYNCcnt
__device__ inline void async_b128(unsigned lds_off, const void* gptr) {
    asm volatile("global_load_async_to_lds_b128 %0, %1, off"
                 :: "v"(lds_off), "v"((unsigned long long)(uintptr_t)gptr)
                 : "memory");
}
__device__ inline void wait_async0() {
    asm volatile("s_wait_asynccnt 0x0" ::: "memory");
}

// ---------------------------------------------------------------------------
// prep: zero norm accumulators, derive alpha scale/shift scalars, and build
// f16 transposed+padded weight buffers laid out as [N][Kpad] (K-pad zeroed;
// this also makes clamped async loads of A's K-pad region harmless).
// ---------------------------------------------------------------------------
__global__ void prep_kernel(const float* __restrict__ Pd, const float* __restrict__ Ps,
                            const float* __restrict__ Pb, const float* __restrict__ Wc,
                            const float* a0, const float* a1, const float* a2, const float* a3,
                            float* __restrict__ norms, float* __restrict__ sc6,
                            _Float16* __restrict__ w0t, _Float16* __restrict__ w1t,
                            _Float16* __restrict__ wct)
{
    size_t gid    = (size_t)blockIdx.x * blockDim.x + threadIdx.x;
    size_t stride = (size_t)gridDim.x * blockDim.x;

    if (gid == 0) { norms[0] = 0.0f; norms[1] = 0.0f; }
    if (gid < 4) {
        const float* as[4] = {a0, a1, a2, a3};
        const int CH1[4] = {0, 1, 1, 1};
        const int CH2[4] = {1, 2, 3, 4};
        int i = (int)gid;
        const float* a = as[i];
        int np = 2 + i;
        float sum1 = 0.f, sum2 = 0.f, sumc = 0.f;
        for (int j = 0; j < np; ++j) { sum1 += a[j]; sum2 += a[np + j]; }
        for (int j = 0; j < 4;  ++j) { sumc += a[2 * np + j]; }
        float s1 = a[CH1[i]], s2 = a[np + CH2[i]], so = a[2 * np + i /*CHK[i]==i*/];
        sc6[i*6 + 0] = s1;  sc6[i*6 + 1] = sum1 - s1;
        sc6[i*6 + 2] = s2;  sc6[i*6 + 3] = sum2 - s2;
        sc6[i*6 + 4] = so;  sc6[i*6 + 5] = sumc - so;
    }

    const size_t W0N = (size_t)400 * 640;
    const size_t W1N = (size_t)400 * 832;
    const size_t WCN = (size_t)4 * 400 * 416;
    for (size_t j = gid; j < W0N + W1N + WCN; j += stride) {
        if (j < W0N) {                       // [Pd(208); Ps(416); pad] -> [n][640]
            int n = (int)(j / 640), k = (int)(j % 640);
            float v = (k < 208) ? Pd[(size_t)k * 400 + n]
                    : (k < 624) ? Ps[(size_t)(k - 208) * 400 + n] : 0.0f;
            w0t[j] = (_Float16)v;
        } else if (j < W0N + W1N) {          // [Ps(416); Pb(400); pad] -> [n][832]
            size_t jj = j - W0N;
            int n = (int)(jj / 832), k = (int)(jj % 832);
            float v = (k < 416) ? Ps[(size_t)k * 400 + n]
                    : (k < 816) ? Pb[(size_t)(k - 416) * 400 + n] : 0.0f;
            w1t[jj] = (_Float16)v;
        } else {                             // Wc[i, i] (400x400) -> [n][416], i=0..3
            size_t jj = j - W0N - W1N;
            int blk = (int)(jj / ((size_t)400 * 416));
            size_t r = jj % ((size_t)400 * 416);
            int n = (int)(r / 416), k = (int)(r % 416);
            float v = (k < 400)
                ? Wc[(((size_t)blk * 5) * 400 + k) * 400 + n]   // (i*4 + CHK[i]=i) == i*5
                : 0.0f;
            wct[jj] = (_Float16)v;
        }
    }
}

// ---------------------------------------------------------------------------
// embed: node01[b] = [dense_emb(208) | sparse_emb(416)] in f16, plus global
// sum-of-squares for the two Frobenius norms. One block per batch row.
// ---------------------------------------------------------------------------
__global__ void embed_kernel(const float* __restrict__ raw_dense,
                             const int* __restrict__ raw_sparse,
                             const float* __restrict__ emb,
                             _Float16* __restrict__ node01,
                             float* __restrict__ norms)
{
    int b = blockIdx.x;
    int tid = threadIdx.x;
    __shared__ float sD, sS;
    if (tid == 0) { sD = 0.f; sS = 0.f; }
    __syncthreads();

    float ld = 0.f, ls = 0.f;
    for (int idx = tid; idx < 624; idx += 256) {
        float v;
        if (idx < 208) {
            int f = idx >> 4, d = idx & 15;
            v = emb[f * 16 + d] * raw_dense[b * ND_ + f];
            ld += v * v;
        } else {
            int j = idx - 208;
            int s = j >> 4, d = j & 15;
            int row = ND_ + VPF_ * s + raw_sparse[b * NS_ + s];
            v = emb[(size_t)row * 16 + d];
            ls += v * v;
        }
        node01[(size_t)b * 624 + idx] = (_Float16)v;
    }
    atomicAdd(&sD, ld);
    atomicAdd(&sS, ls);
    __syncthreads();
    if (tid == 0) { atomicAdd(&norms[0], sD); atomicAdd(&norms[1], sS); }
}

// ---------------------------------------------------------------------------
// WMMA GEMM: C[M x Ncols] = epilogue( concat(s1*A1+c1, s2*A2+c2) @ B )
//   A f16 row-major (two K-segments; boundaries multiples of 8).
//   B pre-transposed f16 [Ncols][Kpad], K-pad rows zeroed.
// Block tile 256x64 (8 waves, 32x64 per wave, 8 accumulators), double-
// buffered LDS with async prefetch of tile i+1 overlapping WMMA of tile i.
// ---------------------------------------------------------------------------
__global__ __launch_bounds__(256) void gemm_wmma_kernel(
    const _Float16* __restrict__ A1, int ldA1, int K1,
    const _Float16* __restrict__ A2, int ldA2, int K2,
    const _Float16* __restrict__ Bt, int Kpad, int Ncols,
    const float* __restrict__ sc,    // {s1,c1,s2,c2} or nullptr
    const float* __restrict__ bias,  // per-col f32 or nullptr
    const float* __restrict__ oc,    // {so,co} or nullptr
    int relu,
    _Float16* __restrict__ C, int ldC)
{
    __shared__ __align__(16) _Float16 Al[2][256 * 40];
    __shared__ __align__(16) _Float16 Bl[2][64 * 40];

    const int tid   = threadIdx.x;
    const int lane  = tid & 31;
    const int wave  = tid >> 5;
    const int lrow  = lane & 15;
    const int lhalf = lane >> 4;
    const int nBase = blockIdx.x * 64;
    const int mBase = blockIdx.y * 256;

    float s1 = 1.f, c1 = 0.f, s2 = 1.f, c2 = 0.f;
    if (sc) { s1 = sc[0]; c1 = sc[1]; s2 = sc[2]; c2 = sc[3]; }
    const int  Ktot  = K1 + K2;
    const bool ident = (s1 == 1.0f) & (c1 == 0.0f) & (s2 == 1.0f) & (c2 == 0.0f);

    v8f acc[8];
#pragma unroll
    for (int g = 0; g < 8; ++g)
#pragma unroll
        for (int v = 0; v < 8; ++v) acc[g][v] = 0.0f;

    // B-tile chunk owned by this thread (fixed across k-steps)
    const int bn  = tid >> 2;
    const int bkl = (tid & 3) * 8;
    const int bgn = (nBase + bn < Ncols) ? (nBase + bn) : (Ncols - 1); // clamp: junk cols discarded

    // stage tile at k-offset kb into LDS buffer `buf` (async fast path)
    auto stage_fast = [&](int kb, int buf) {
        async_b128((unsigned)(uintptr_t)&Bl[buf][bn * 40 + bkl],
                   Bt + (size_t)bgn * Kpad + kb + bkl);
#pragma unroll
        for (int t = 0; t < 4; ++t) {
            int ch = tid + t * 256;
            int r  = ch >> 2;
            int kl = (ch & 3) * 8;
            int kg = kb + kl;
            const _Float16* src;
            if (kg < K1)        src = A1 + (size_t)(mBase + r) * ldA1 + kg;
            else if (kg < Ktot) src = A2 + (size_t)(mBase + r) * ldA2 + (kg - K1);
            else                src = A1 + (size_t)(mBase + r) * ldA1; // clamp (B pad is 0)
            async_b128((unsigned)(uintptr_t)&Al[buf][r * 40 + kl], src);
        }
    };
    // general path: elementwise x = s*node + c while filling LDS
    auto stage_slow = [&](int kb, int buf) {
        async_b128((unsigned)(uintptr_t)&Bl[buf][bn * 40 + bkl],
                   Bt + (size_t)bgn * Kpad + kb + bkl);
#pragma unroll
        for (int t = 0; t < 4; ++t) {
            int ch = tid + t * 256;
            int r  = ch >> 2;
            int kl = (ch & 3) * 8;
            int kg = kb + kl;
            h8 v = h8zero();
            if (kg < Ktot) {
                const _Float16* src; float s, cc;
                if (kg < K1) { src = A1 + (size_t)(mBase + r) * ldA1 + kg;        s = s1; cc = c1; }
                else         { src = A2 + (size_t)(mBase + r) * ldA2 + (kg - K1); s = s2; cc = c2; }
                h8 raw = *(const h8*)src;
#pragma unroll
                for (int j = 0; j < 8; ++j) v[j] = (_Float16)(s * (float)raw[j] + cc);
            }
            *(h8*)&Al[buf][r * 40 + kl] = v;
        }
    };
    // 8 WMMAs on the tile sitting in buffer `buf`
    auto compute = [&](int buf) {
        const int arow0 = wave * 32 + lrow;
        h8 a0lo = *(const h8*)&Al[buf][arow0 * 40 + lhalf * 8];
        h8 a0hi = *(const h8*)&Al[buf][arow0 * 40 + lhalf * 8 + 16];
        h8 a1lo = *(const h8*)&Al[buf][(arow0 + 16) * 40 + lhalf * 8];
        h8 a1hi = *(const h8*)&Al[buf][(arow0 + 16) * 40 + lhalf * 8 + 16];
        v16h am0 = __builtin_shufflevector(a0lo, a0hi, 0,1,2,3,4,5,6,7,8,9,10,11,12,13,14,15);
        v16h am1 = __builtin_shufflevector(a1lo, a1hi, 0,1,2,3,4,5,6,7,8,9,10,11,12,13,14,15);
#pragma unroll
        for (int f = 0; f < 4; ++f) {
            int bcol = f * 16 + lrow;
            h8 blo = *(const h8*)&Bl[buf][bcol * 40 + lhalf * 8];
            h8 bhi = *(const h8*)&Bl[buf][bcol * 40 + lhalf * 8 + 16];
            v16h bb = __builtin_shufflevector(blo, bhi, 0,1,2,3,4,5,6,7,8,9,10,11,12,13,14,15);
            acc[f]     = __builtin_amdgcn_wmma_f32_16x16x32_f16(
                false, am0, false, bb, (short)0, acc[f],     false, false);
            acc[4 + f] = __builtin_amdgcn_wmma_f32_16x16x32_f16(
                false, am1, false, bb, (short)0, acc[4 + f], false, false);
        }
    };

    const int nk = Kpad >> 5;   // k-steps of 32; Kpad is a multiple of 32
    if (ident) {
        stage_fast(0, 0);
        wait_async0();
        __syncthreads();
        for (int i = 0; i < nk; ++i) {
            int cur = i & 1;
            if (i + 1 < nk) stage_fast((i + 1) << 5, cur ^ 1);  // prefetch overlaps WMMA
            compute(cur);
            wait_async0();
            __syncthreads();
        }
    } else {
        stage_slow(0, 0);
        wait_async0();
        __syncthreads();
        for (int i = 0; i < nk; ++i) {
            int cur = i & 1;
            if (i + 1 < nk) stage_slow((i + 1) << 5, cur ^ 1);
            compute(cur);
            wait_async0();
            __syncthreads();
        }
    }

    float so = 1.f, co = 0.f;
    if (oc) { so = oc[0]; co = oc[1]; }

    // C layout: lanes 0-15 -> n=lane, M=v; lanes 16-31 -> n=lane-16, M=v+8
#pragma unroll
    for (int mi = 0; mi < 2; ++mi) {
#pragma unroll
        for (int f = 0; f < 4; ++f) {
            int col = nBase + f * 16 + lrow;
            if (col < Ncols) {
                float bval = bias ? bias[col] : 0.0f;
#pragma unroll
                for (int v = 0; v < 8; ++v) {
                    int row = mBase + wave * 32 + mi * 16 + lhalf * 8 + v;
                    float x = acc[mi * 4 + f][v] + bval;
                    if (relu) x = fmaxf(x, 0.0f);
                    x = so * x + co;
                    C[(size_t)row * ldC + col] = (_Float16)x;
                }
            }
        }
    }
}

// ---------------------------------------------------------------------------
// classifier: logits[b] = nodes[b,0:1600] . clf_w + clf_b. One wave per row.
// Also emits regs = 1e-5*(||dense||_F + ||sparse||_F) at out[B].
// ---------------------------------------------------------------------------
__global__ void classifier_kernel(const _Float16* __restrict__ nodes,
                                  const float* __restrict__ w,
                                  const float* __restrict__ bptr,
                                  const float* __restrict__ norms,
                                  float* __restrict__ out)
{
    int tid  = threadIdx.x;
    int lane = tid & 31;
    int wave = tid >> 5;
    int row  = blockIdx.x * 8 + wave;

    const _Float16* nr = nodes + (size_t)row * 1600;
    float sum = 0.0f;
    for (int ch = lane; ch < 200; ch += 32) {     // 200 chunks of 8
        h8 v = *(const h8*)(nr + ch * 8);
        const float* wp = w + ch * 8;
#pragma unroll
        for (int j = 0; j < 8; ++j) sum += (float)v[j] * wp[j];
    }
#pragma unroll
    for (int off = 16; off > 0; off >>= 1) sum += __shfl_xor(sum, off, 32);
    if (lane == 0) out[row] = sum + bptr[0];

    if (blockIdx.x == 0 && tid == 0)
        out[B_] = 1e-5f * (sqrtf(norms[0]) + sqrtf(norms[1]));
}

// ---------------------------------------------------------------------------
extern "C" void kernel_launch(void* const* d_in, const int* in_sizes, int n_in,
                              void* d_out, int out_size, void* d_ws, size_t ws_size,
                              hipStream_t stream)
{
    const float* raw_dense  = (const float*)d_in[0];
    const int*   raw_sparse = (const int*)  d_in[1];
    const float* emb        = (const float*)d_in[2];
    const float* Pd         = (const float*)d_in[3];
    const float* Ps         = (const float*)d_in[4];
    const float* Pb         = (const float*)d_in[5];
    const float* Wc         = (const float*)d_in[6];
    const float* bc         = (const float*)d_in[7];
    const float* clfw       = (const float*)d_in[8];
    const float* clfb       = (const float*)d_in[9];
    const float* a0         = (const float*)d_in[10];
    const float* a1         = (const float*)d_in[11];
    const float* a2         = (const float*)d_in[12];
    const float* a3         = (const float*)d_in[13];

    char* ws = (char*)d_ws;                       // requires ws_size >= WS_END (~45.5 MB)
    float*     norms  = (float*)(ws + OFF_NORM);
    float*     sc6    = (float*)(ws + OFF_SC);
    _Float16*  w0t    = (_Float16*)(ws + OFF_W0T);
    _Float16*  w1t    = (_Float16*)(ws + OFF_W1T);
    _Float16*  wct    = (_Float16*)(ws + OFF_WCT);
    _Float16*  node01 = (_Float16*)(ws + OFF_N01);
    _Float16*  h      = (_Float16*)(ws + OFF_H);
    _Float16*  nodes  = (_Float16*)(ws + OFF_NODES);
    float*     out    = (float*)d_out;

    prep_kernel<<<2048, 256, 0, stream>>>(Pd, Ps, Pb, Wc, a0, a1, a2, a3,
                                          norms, sc6, w0t, w1t, wct);
    embed_kernel<<<B_, 256, 0, stream>>>(raw_dense, raw_sparse, emb, node01, norms);

    dim3 ggrid(7, 32);   // ceil(400/64) x (8192/256)

    // block 0: h = relu([dense|sparse] @ W0), node2 = h @ Wc0 + bc0
    gemm_wmma_kernel<<<ggrid, 256, 0, stream>>>(
        node01, 624, 208, node01 + 208, 624, 416,
        w0t, 640, BIN_, sc6 + 0, nullptr, nullptr, 1, h, BOUT_);
    gemm_wmma_kernel<<<ggrid, 256, 0, stream>>>(
        h, 400, 400, h, 400, 0,
        wct, 416, BOUT_, nullptr, bc + (size_t)(0 * 5) * BOUT_, sc6 + 4, 0,
        nodes + 0, 1600);

    // blocks 1..3: h = relu([sparse|node_{i+1}] @ W1), node = h @ Wc_i + bc_i
    for (int i = 1; i < 4; ++i) {
        gemm_wmma_kernel<<<ggrid, 256, 0, stream>>>(
            node01 + 208, 624, 416, nodes + (size_t)(i - 1) * 400, 1600, 400,
            w1t, 832, BIN_, sc6 + i * 6, nullptr, nullptr, 1, h, BOUT_);
        gemm_wmma_kernel<<<ggrid, 256, 0, stream>>>(
            h, 400, 400, h, 400, 0,
            wct + (size_t)i * 400 * 416, 416, BOUT_,
            nullptr, bc + (size_t)(i * 5) * BOUT_, sc6 + i * 6 + 4, 0,
            nodes + (size_t)i * 400, 1600);
    }

    classifier_kernel<<<B_ / 8, 256, 0, stream>>>(nodes, clfw, clfb, norms, out);
}
